// MultiHead_Attention_19739669692947
// MI455X (gfx1250) — matrix operand order repaired and checked by reading, not compiled, not run on hardware
//
#include <hip/hip_runtime.h>

// MI455X / gfx1250, wave32. All matmuls on v_wmma_f32_16x16x32_bf16.
typedef __bf16 bf16_t;
typedef __attribute__((ext_vector_type(16))) __bf16 v16bf;
typedef __attribute__((ext_vector_type(8)))  __bf16 v8bf;
typedef __attribute__((ext_vector_type(8)))  float  v8f;

#define D_MODEL 1024
#define NSEQ    2048
#define NHEAD   16
#define HEADD   64
#define BATCH   2
#define NROWS   (BATCH * NSEQ)   // 4096

__device__ __forceinline__ v8f wmma_bf16(v16bf a, v16bf b, v8f c) {
  // (neg_a, A, neg_b, B, c_mod, C, reuse_a, reuse_b)
  return __builtin_amdgcn_wmma_f32_16x16x32_bf16(false, a, false, b, (short)0, c, false, false);
}

__device__ __forceinline__ v8f vzero8() {
  v8f z;
#pragma unroll
  for (int i = 0; i < 8; ++i) z[i] = 0.0f;
  return z;
}

// A fragment: 16(M) x 32(K) bf16, source row-major with leading dim ld.
// Lane layout (ISA 7.12.2): m = lane&15, elems 0..7 -> K = half*8+0..7,
// elems 8..15 -> K = 16 + half*8 + 0..7.
__device__ __forceinline__ v16bf ldA(const bf16_t* tile, int ld, int lane) {
  const int m = lane & 15, h = (lane >> 4) & 1;
  const bf16_t* p = tile + (size_t)m * ld + h * 8;
  union { v16bf v; v8bf q[2]; } u;
  u.q[0] = *(const v8bf*)(p);
  u.q[1] = *(const v8bf*)(p + 16);
  return u.v;
}

// B fragment: 32(K) x 16(N) bf16 taken from a [N, K] row-major buffer (i.e.
// the transposed operand), leading dim ld. Lane layout: n = lane&15,
// elems 0..7 -> K = half*8 + 0..7, elems 8..15 -> K = 16 + half*8 + 0..7.
__device__ __forceinline__ v16bf ldB(const bf16_t* tileT, int ld, int lane) {
  const int n = lane & 15, h = (lane >> 4) & 1;
  const bf16_t* p = tileT + (size_t)n * ld + h * 8;
  union { v16bf v; v8bf q[2]; } u;
  u.q[0] = *(const v8bf*)(p);
  u.q[1] = *(const v8bf*)(p + 16);
  return u.v;
}

__global__ __launch_bounds__(256) void cvt_f32_bf16(const float* __restrict__ in,
                                                    bf16_t* __restrict__ out, int n) {
  int i = blockIdx.x * blockDim.x + threadIdx.x;
  if (i < n) out[i] = (bf16_t)in[i];
}

// W is [in=K, out=N] row-major f32; produce WT [N, K] row-major bf16.
__global__ __launch_bounds__(256) void cvt_transpose_w(const float* __restrict__ W,
                                                       bf16_t* __restrict__ WT) {
  int i = blockIdx.x * blockDim.x + threadIdx.x;  // over D*D
  int k = i >> 10;            // row of W
  int n = i & (D_MODEL - 1);  // col of W
  WT[(size_t)n * D_MODEL + k] = (bf16_t)W[i];
}

// Fused QKV projection. Q,K stored [B,H,N,HD]; V stored transposed [B,H,HD,N].
__global__ __launch_bounds__(256) void qkv_gemm(
    const bf16_t* __restrict__ xb,
    const bf16_t* __restrict__ WqT, const bf16_t* __restrict__ WkT,
    const bf16_t* __restrict__ WvT,
    bf16_t* __restrict__ Q, bf16_t* __restrict__ K, bf16_t* __restrict__ VT) {
  const int lane = threadIdx.x & 31;
  const int wave = threadIdx.x >> 5;
  const int wm = wave & 3, wn = wave >> 2;      // 4 x 2 wave grid
  const int row0 = blockIdx.x * 128 + wm * 32;  // rows of x (B*N flat)
  const int col0 = blockIdx.y * 64 + wn * 32;   // cols of projection

  v8f aq[2][2], ak[2][2], av[2][2];
#pragma unroll
  for (int i = 0; i < 2; ++i)
#pragma unroll
    for (int j = 0; j < 2; ++j) { aq[i][j] = vzero8(); ak[i][j] = vzero8(); av[i][j] = vzero8(); }

  for (int k = 0; k < D_MODEL; k += 32) {
    v16bf a0 = ldA(xb + (size_t)row0 * D_MODEL + k, D_MODEL, lane);
    v16bf a1 = ldA(xb + (size_t)(row0 + 16) * D_MODEL + k, D_MODEL, lane);
    {
      v16bf b0 = ldB(WqT + (size_t)col0 * D_MODEL + k, D_MODEL, lane);
      v16bf b1 = ldB(WqT + (size_t)(col0 + 16) * D_MODEL + k, D_MODEL, lane);
      aq[0][0] = wmma_bf16(a0, b0, aq[0][0]);
      aq[0][1] = wmma_bf16(a0, b1, aq[0][1]);
      aq[1][0] = wmma_bf16(a1, b0, aq[1][0]);
      aq[1][1] = wmma_bf16(a1, b1, aq[1][1]);
    }
    {
      v16bf b0 = ldB(WkT + (size_t)col0 * D_MODEL + k, D_MODEL, lane);
      v16bf b1 = ldB(WkT + (size_t)(col0 + 16) * D_MODEL + k, D_MODEL, lane);
      ak[0][0] = wmma_bf16(a0, b0, ak[0][0]);
      ak[0][1] = wmma_bf16(a0, b1, ak[0][1]);
      ak[1][0] = wmma_bf16(a1, b0, ak[1][0]);
      ak[1][1] = wmma_bf16(a1, b1, ak[1][1]);
    }
    {
      v16bf b0 = ldB(WvT + (size_t)col0 * D_MODEL + k, D_MODEL, lane);
      v16bf b1 = ldB(WvT + (size_t)(col0 + 16) * D_MODEL + k, D_MODEL, lane);
      av[0][0] = wmma_bf16(a0, b0, av[0][0]);
      av[0][1] = wmma_bf16(a0, b1, av[0][1]);
      av[1][0] = wmma_bf16(a1, b0, av[1][0]);
      av[1][1] = wmma_bf16(a1, b1, av[1][1]);
    }
  }

  const int half = (lane >> 4) & 1;
#pragma unroll
  for (int mi = 0; mi < 2; ++mi)
#pragma unroll
    for (int ni = 0; ni < 2; ++ni) {
      const int colT = col0 + ni * 16 + (lane & 15);
      const int h = colT >> 6, hd = colT & 63;
      const int rbase = row0 + mi * 16 + half * 8;  // 8 consecutive rows, same batch
      const int b = rbase / NSEQ;
      const int seq0 = rbase & (NSEQ - 1);
      const size_t qk_base = (((size_t)(b * NHEAD + h)) * NSEQ + seq0) * HEADD + hd;
#pragma unroll
      for (int r = 0; r < 8; ++r) {
        Q[qk_base + (size_t)r * HEADD] = (bf16_t)aq[mi][ni][r];
        K[qk_base + (size_t)r * HEADD] = (bf16_t)ak[mi][ni][r];
      }
      v8bf vt;
#pragma unroll
      for (int r = 0; r < 8; ++r) vt[r] = (bf16_t)av[mi][ni][r];
      const size_t vt_base = (((size_t)(b * NHEAD + h)) * HEADD + hd) * NSEQ + seq0;
      *(v8bf*)(VT + vt_base) = vt;  // contiguous along seq
    }
}

// Causal flash attention. One wave = 16 query rows; 8 waves/block = 128 rows.
// grid = (NSEQ/128, B*H). ctx written bf16 as [B*N, D_MODEL].
__global__ __launch_bounds__(256) void flash_attn(
    const bf16_t* __restrict__ Q, const bf16_t* __restrict__ K,
    const bf16_t* __restrict__ VT, bf16_t* __restrict__ ctx) {
  __shared__ __align__(16) bf16_t pbuf[8][16 * 32];  // per-wave P staging (C->A relayout)
  const int lane = threadIdx.x & 31;
  const int wave = threadIdx.x >> 5;
  const int half = (lane >> 4) & 1;
  const int nloc = lane & 15;
  const int bh = blockIdx.y;
  const int q0 = blockIdx.x * 128 + wave * 16;

  const bf16_t* Qb = Q + (size_t)bh * NSEQ * HEADD;
  const bf16_t* Kb = K + (size_t)bh * NSEQ * HEADD;
  const bf16_t* Vb = VT + (size_t)bh * HEADD * NSEQ;

  const v16bf qa0 = ldA(Qb + (size_t)q0 * HEADD, HEADD, lane);       // hd 0..31
  const v16bf qa1 = ldA(Qb + (size_t)q0 * HEADD + 32, HEADD, lane);  // hd 32..63

  v8f acc[4];
#pragma unroll
  for (int j = 0; j < 4; ++j) acc[j] = vzero8();
  float rowm[8], rowl[8];
#pragma unroll
  for (int r = 0; r < 8; ++r) { rowm[r] = -3.0e38f; rowl[r] = 0.0f; }

  const float cscale = 0.125f * 1.44269504f;  // (1/sqrt(HD)) * log2(e)
  const int nkt = (q0 + 16 + 31) >> 5;        // key tiles of 32 covering keys <= q0+15

  for (int kt = 0; kt < nkt; ++kt) {
    const int k0 = kt * 32;
    v8f s0 = vzero8(), s1 = vzero8();
    s0 = wmma_bf16(qa0, ldB(Kb + (size_t)k0 * HEADD, HEADD, lane), s0);
    s0 = wmma_bf16(qa1, ldB(Kb + (size_t)k0 * HEADD + 32, HEADD, lane), s0);
    s1 = wmma_bf16(qa0, ldB(Kb + (size_t)(k0 + 16) * HEADD, HEADD, lane), s1);
    s1 = wmma_bf16(qa1, ldB(Kb + (size_t)(k0 + 16) * HEADD + 32, HEADD, lane), s1);

#pragma unroll
    for (int r = 0; r < 8; ++r) {
      const int qi = q0 + half * 8 + r;
      float v0 = s0[r] * cscale;
      float v1 = s1[r] * cscale;
      if (k0 + nloc > qi)      v0 = -3.0e38f;  // causal mask
      if (k0 + 16 + nloc > qi) v1 = -3.0e38f;
      float tmax = fmaxf(v0, v1);
#pragma unroll
      for (int off = 1; off < 16; off <<= 1)
        tmax = fmaxf(tmax, __shfl_xor(tmax, off, 32));
      const float newm = fmaxf(rowm[r], tmax);
      const float rescale = exp2f(rowm[r] - newm);
      rowm[r] = newm;
      const float p0 = exp2f(v0 - newm);
      const float p1 = exp2f(v1 - newm);
      float ps = p0 + p1;
#pragma unroll
      for (int off = 1; off < 16; off <<= 1)
        ps += __shfl_xor(ps, off, 32);
      rowl[r] = rowl[r] * rescale + ps;
#pragma unroll
      for (int j = 0; j < 4; ++j) acc[j][r] *= rescale;
      const int prow = half * 8 + r;
      pbuf[wave][prow * 32 + nloc]      = (bf16_t)p0;
      pbuf[wave][prow * 32 + 16 + nloc] = (bf16_t)p1;
    }
    // wave-local RAW on LDS: stores above -> ldA below (no block barrier needed)
    asm volatile("s_wait_dscnt 0" ::: "memory");
    const v16bf pa = ldA(&pbuf[wave][0], 32, lane);
#pragma unroll
    for (int j = 0; j < 4; ++j)
      acc[j] = wmma_bf16(pa, ldB(Vb + (size_t)(j * 16) * NSEQ + k0, NSEQ, lane), acc[j]);
  }

  const int b = bh >> 4, h = bh & 15;
#pragma unroll
  for (int j = 0; j < 4; ++j) {
    const int col = h * HEADD + j * 16 + nloc;
#pragma unroll
    for (int r = 0; r < 8; ++r) {
      const int seq = q0 + half * 8 + r;
      ctx[((size_t)(b * NSEQ + seq)) * D_MODEL + col] = (bf16_t)(acc[j][r] / rowl[r]);
    }
  }
}

// out = ctx(bf16) @ Wo, f32 output.
__global__ __launch_bounds__(256) void out_gemm(const bf16_t* __restrict__ ctx,
                                                const bf16_t* __restrict__ WoT,
                                                float* __restrict__ out) {
  const int lane = threadIdx.x & 31;
  const int wave = threadIdx.x >> 5;
  const int wm = wave & 3, wn = wave >> 2;
  const int row0 = blockIdx.x * 128 + wm * 32;
  const int col0 = blockIdx.y * 64 + wn * 32;

  v8f acc[2][2];
#pragma unroll
  for (int i = 0; i < 2; ++i)
#pragma unroll
    for (int j = 0; j < 2; ++j) acc[i][j] = vzero8();

  for (int k = 0; k < D_MODEL; k += 32) {
    v16bf a0 = ldA(ctx + (size_t)row0 * D_MODEL + k, D_MODEL, lane);
    v16bf a1 = ldA(ctx + (size_t)(row0 + 16) * D_MODEL + k, D_MODEL, lane);
    v16bf b0 = ldB(WoT + (size_t)col0 * D_MODEL + k, D_MODEL, lane);
    v16bf b1 = ldB(WoT + (size_t)(col0 + 16) * D_MODEL + k, D_MODEL, lane);
    acc[0][0] = wmma_bf16(a0, b0, acc[0][0]);
    acc[0][1] = wmma_bf16(a0, b1, acc[0][1]);
    acc[1][0] = wmma_bf16(a1, b0, acc[1][0]);
    acc[1][1] = wmma_bf16(a1, b1, acc[1][1]);
  }

  const int half = (lane >> 4) & 1;
#pragma unroll
  for (int mi = 0; mi < 2; ++mi)
#pragma unroll
    for (int ni = 0; ni < 2; ++ni) {
      const int col = col0 + ni * 16 + (lane & 15);
      const int r0 = row0 + mi * 16 + half * 8;
#pragma unroll
      for (int r = 0; r < 8; ++r)
        out[(size_t)(r0 + r) * D_MODEL + col] = acc[mi][ni][r];
    }
}

extern "C" void kernel_launch(void* const* d_in, const int* in_sizes, int n_in,
                              void* d_out, int out_size, void* d_ws, size_t ws_size,
                              hipStream_t stream) {
  const float* x  = (const float*)d_in[0];
  const float* Wq = (const float*)d_in[1];
  const float* Wk = (const float*)d_in[2];
  const float* Wv = (const float*)d_in[3];
  const float* Wo = (const float*)d_in[4];

  char* ws = (char*)d_ws;
  size_t off = 0;
  const size_t XB  = (size_t)NROWS * D_MODEL * sizeof(bf16_t);                 // 8 MB
  const size_t WB  = (size_t)D_MODEL * D_MODEL * sizeof(bf16_t);               // 2 MB
  const size_t QKB = (size_t)BATCH * NHEAD * NSEQ * HEADD * sizeof(bf16_t);    // 8 MB

  bf16_t* xb  = (bf16_t*)(ws + off); off += XB;
  bf16_t* WqT = (bf16_t*)(ws + off); off += WB;
  bf16_t* WkT = (bf16_t*)(ws + off); off += WB;
  bf16_t* WvT = (bf16_t*)(ws + off); off += WB;
  bf16_t* WoT = (bf16_t*)(ws + off); off += WB;
  bf16_t* Qm  = (bf16_t*)(ws + off); off += QKB;
  bf16_t* Km  = (bf16_t*)(ws + off); off += QKB;
  bf16_t* VTm = (bf16_t*)(ws + off); off += QKB;
  bf16_t* ctx = (bf16_t*)(ws + off); off += XB;   // ~48 MB total

  cvt_f32_bf16<<<(NROWS * D_MODEL) / 256, 256, 0, stream>>>(x, xb, NROWS * D_MODEL);
  cvt_transpose_w<<<(D_MODEL * D_MODEL) / 256, 256, 0, stream>>>(Wq, WqT);
  cvt_transpose_w<<<(D_MODEL * D_MODEL) / 256, 256, 0, stream>>>(Wk, WkT);
  cvt_transpose_w<<<(D_MODEL * D_MODEL) / 256, 256, 0, stream>>>(Wv, WvT);
  cvt_transpose_w<<<(D_MODEL * D_MODEL) / 256, 256, 0, stream>>>(Wo, WoT);

  qkv_gemm<<<dim3(NROWS / 128, D_MODEL / 64), 256, 0, stream>>>(xb, WqT, WkT, WvT, Qm, Km, VTm);
  flash_attn<<<dim3(NSEQ / 128, BATCH * NHEAD), 256, 0, stream>>>(Qm, Km, VTm, ctx);
  out_gemm<<<dim3(NROWS / 128, D_MODEL / 64), 256, 0, stream>>>(ctx, WoT, (float*)d_out);
}
